// PointSelfAttention_84524956385318
// MI455X (gfx1250) — compile-verified, matrix operand order checked by reading, not run on hardware
//
#include <hip/hip_runtime.h>

#define KPTS   64
#define HID    512
#define NH     8
#define HD     64
#define RBF    32
#define QKV3   1536
#define NTQ    96      // 1536/16 N-tiles for qkv weight
#define NTO    32      // 512/16  N-tiles for out weight
#define KTH    16      // 512/32  K-steps over hidden

typedef __attribute__((ext_vector_type(16))) __bf16 bf16x16;
typedef __attribute__((ext_vector_type(8)))  __bf16 bf16x8;
typedef __attribute__((ext_vector_type(4)))  __bf16 bf16x4;
typedef __attribute__((ext_vector_type(8)))  float  v8f;

__device__ __forceinline__ v8f wmma_bf16(bf16x16 a, bf16x16 b, v8f c) {
  // D = A(16x32 bf16) * B(32x16 bf16) + C(16x16 f32)
  return __builtin_amdgcn_wmma_f32_16x16x32_bf16(false, a, false, b, (short)0, c, false, false);
}

// Load an A-style fragment (also used for B fed as rows of the transposed source)
// from a row-major bf16 matrix. Lane l holds row (row0 + l%16); its 16 bf16 are
// k = {k0..k0+7} and {16+k0..16+k0+7} with k0 = 8*(l>=16), per ISA 7.12.2.
__device__ __forceinline__ bf16x16 load_frag(const __bf16* base, int row0, int stride, int kt, int lane) {
  const __bf16* p = base + (row0 + (lane & 15)) * stride + kt * 32 + ((lane >> 4) << 3);
  bf16x8 lo = *(const bf16x8*)(p);
  bf16x8 hi = *(const bf16x8*)(p + 16);
  return __builtin_shufflevector(lo, hi, 0,1,2,3,4,5,6,7,8,9,10,11,12,13,14,15);
}

// Pre-swizzled weight fragments: one contiguous 32B chunk per lane per tile.
__device__ __forceinline__ bf16x16 load_wfrag(const __bf16* w, int ntiles, int kt, int nt, int lane) {
  return *(const bf16x16*)(w + (((kt * ntiles + nt) * 32 + lane) << 4));
}

// ---------------------------------------------------------------------------
// Kernel 1: reformat w_qkv (512x1536) and w_out (512x512) f32 -> bf16 B-fragments.
// ---------------------------------------------------------------------------
__global__ void prep_weights(const float* __restrict__ wq_src, const float* __restrict__ wo_src,
                             __bf16* __restrict__ wq, __bf16* __restrict__ wo) {
  const int NQ = KTH * NTQ * 512;   // 786432 elements
  const int NO = KTH * NTO * 512;   // 262144 elements
  int idx = blockIdx.x * blockDim.x + threadIdx.x;
  if (idx < NQ) {
    int e = idx & 15, lane = (idx >> 4) & 31, tile = idx >> 9;
    int nt = tile % NTQ, kt = tile / NTQ;
    int k = kt * 32 + ((e & 8) << 1) + ((lane >> 4) << 3) + (e & 7);
    int n = nt * 16 + (lane & 15);
    wq[idx] = (__bf16)wq_src[k * QKV3 + n];
  } else if (idx < NQ + NO) {
    int f = idx - NQ;
    int e = f & 15, lane = (f >> 4) & 31, tile = f >> 9;
    int nt = tile & (NTO - 1), kt = tile >> 5;
    int k = kt * 32 + ((e & 8) << 1) + ((lane >> 4) << 3) + (e & 7);
    int n = nt * 16 + (lane & 15);
    wo[f] = (__bf16)wo_src[k * HID + n];
  }
}

// ---------------------------------------------------------------------------
// Kernel 2: fully fused per-molecule attention block. 1 block = 1 molecule.
// 256 threads = 8 waves. All intermediates live in 320KB LDS.
// ---------------------------------------------------------------------------
__global__ __launch_bounds__(256, 1)
void fused_attn(const float* __restrict__ h_in, const float* __restrict__ pos,
                const float* __restrict__ centers, const float* __restrict__ w_rbf,
                const float* __restrict__ b_rbf,
                const __bf16* __restrict__ wq_frag, const float* __restrict__ b_qkv,
                const __bf16* __restrict__ wo_frag, const float* __restrict__ b_out,
                const float* __restrict__ gamma, const float* __restrict__ beta,
                float* __restrict__ out)
{
  __shared__ __align__(16) unsigned char smem[327680];   // 320 KB = full WGP LDS
  // Region B (64KB): RBF bias bf16 [8][64][64] -> later per-head softmax P
  __bf16* BIAS = (__bf16*)smem;
  // Region C (64KB): centers/w_rbf staging -> h bf16 [64][512] -> agg bf16 [64][512]
  __bf16* HB   = (__bf16*)(smem + 65536);
  float*  CEN  = (float*)(smem + 65536);
  float*  WRBF = CEN + RBF;
  // Region D (192KB): pos -> Q/K/Vt bf16 [8][64][64] x3 -> y f32 [64][512]
  unsigned char* regD = smem + 131072;
  float*  POSS = (float*)regD;
  __bf16* QS = (__bf16*)regD;
  __bf16* KS = QS + NH * KPTS * HD;
  __bf16* VT = KS + NH * KPTS * HD;
  float*  Y  = (float*)regD;

  const int tid  = threadIdx.x;
  const int lane = tid & 31;
  const int wave = tid >> 5;
  const int mol  = blockIdx.x;
  const int n0   = lane & 15;
  const int rowoff = (lane >> 4) << 3;

  // ---- stage pos / centers / w_rbf ----
  if (tid < KPTS * 3) POSS[tid] = pos[mol * KPTS * 3 + tid];
  if (tid < RBF)      CEN[tid]  = centers[tid];
  if (tid < RBF * NH) WRBF[tid] = w_rbf[tid];
  __syncthreads();

  // ---- phase 0: RBF attention bias (shared across heads, projected to 8 heads) ----
  {
    const float inv_width = 16.0f;   // 1 / (CUTOFF/NUM_RBF)^2 = 1/0.0625
    float brbf[NH];
    #pragma unroll
    for (int hh = 0; hh < NH; ++hh) brbf[hh] = b_rbf[hh];
    for (int p = tid; p < KPTS * KPTS; p += 256) {
      int i = p >> 6, j = p & 63;
      float dx = POSS[i*3+0] - POSS[j*3+0];
      float dy = POSS[i*3+1] - POSS[j*3+1];
      float dz = POSS[i*3+2] - POSS[j*3+2];
      float d = sqrtf(fmaxf(dx*dx + dy*dy + dz*dz, 1e-12f));
      float acc[NH];
      #pragma unroll
      for (int hh = 0; hh < NH; ++hh) acc[hh] = brbf[hh];
      for (int r = 0; r < RBF; ++r) {
        float t = d - CEN[r];
        float e = __expf(-t * t * inv_width);
        #pragma unroll
        for (int hh = 0; hh < NH; ++hh) acc[hh] += e * WRBF[r * NH + hh];
      }
      #pragma unroll
      for (int hh = 0; hh < NH; ++hh) BIAS[(hh * KPTS + i) * KPTS + j] = (__bf16)acc[hh];
    }
  }
  __syncthreads();   // bias done; CEN/WRBF/POSS now dead

  // ---- phase 1: stage h (f32 global -> bf16 LDS) ----
  const float* hsrc = h_in + (size_t)mol * KPTS * HID;
  for (int t = tid; t < KPTS * HID / 4; t += 256) {
    float4 v = ((const float4*)hsrc)[t];
    bf16x4 b4; b4[0] = (__bf16)v.x; b4[1] = (__bf16)v.y; b4[2] = (__bf16)v.z; b4[3] = (__bf16)v.w;
    *(bf16x4*)(HB + 4 * t) = b4;
  }
  __syncthreads();

  // ---- phase 2: QKV GEMM: [64x512] @ [512x1536], wave owns 12 N-tiles ----
  const float scale = 0.125f;   // D^-0.5, folded into Q
  for (int ntp = 0; ntp < 6; ++ntp) {
    int nt0 = wave * 12 + ntp * 2;
    v8f acc[2][4] = {};
    for (int kt = 0; kt < KTH; ++kt) {
      if (kt + 1 < KTH)
        __builtin_prefetch(wq_frag + ((((kt + 1) * NTQ + nt0) * 32 + lane) << 4), 0, 1);
      bf16x16 b0 = load_wfrag(wq_frag, NTQ, kt, nt0,     lane);
      bf16x16 b1 = load_wfrag(wq_frag, NTQ, kt, nt0 + 1, lane);
      #pragma unroll
      for (int mt = 0; mt < 4; ++mt) {
        bf16x16 a = load_frag(HB, mt * 16, HID, kt, lane);
        acc[0][mt] = wmma_bf16(a, b0, acc[0][mt]);
        acc[1][mt] = wmma_bf16(a, b1, acc[1][mt]);
      }
    }
    #pragma unroll
    for (int u = 0; u < 2; ++u) {
      int c = (nt0 + u) * 16 + n0;     // column in 0..1535 (per-lane, uniform routing per tile)
      float bq = b_qkv[c];
      #pragma unroll
      for (int mt = 0; mt < 4; ++mt) {
        #pragma unroll
        for (int r = 0; r < 8; ++r) {
          int row = mt * 16 + r + rowoff;
          float val = acc[u][mt][r] + bq;
          if (c < 512) {
            QS[(((c >> 6) * KPTS) + row) * HD + (c & 63)] = (__bf16)(val * scale);
          } else if (c < 1024) {
            int cc = c - 512;
            KS[(((cc >> 6) * KPTS) + row) * HD + (cc & 63)] = (__bf16)val;
          } else {
            int cc = c - 1024;
            VT[(((cc >> 6) * HD) + (cc & 63)) * KPTS + row] = (__bf16)val;   // transposed [d][j]
          }
        }
      }
    }
  }
  __syncthreads();   // Q/K/Vt complete; HB (h) dead -> becomes AGG

  // ---- phase 3: per-head attention, wave = head ----
  {
    const int hh = wave;
    const __bf16* Qh = QS + hh * KPTS * HD;
    const __bf16* Kh = KS + hh * KPTS * HD;
    const __bf16* Vh = VT + hh * KPTS * HD;       // [d][j]
    const __bf16* Bh = BIAS + hh * KPTS * KPTS;   // bias for this head
    __bf16* Ph = BIAS + hh * KPTS * KPTS;         // reuse as P (wave-local)
    __bf16* AGG = HB;

    // S = scale*Q @ K^T  (scale already in Q)
    v8f s[4][4] = {};
    #pragma unroll
    for (int kt = 0; kt < 2; ++kt) {
      bf16x16 bfr[4];
      #pragma unroll
      for (int nt = 0; nt < 4; ++nt) bfr[nt] = load_frag(Kh, nt * 16, HD, kt, lane);
      #pragma unroll
      for (int mt = 0; mt < 4; ++mt) {
        bf16x16 a = load_frag(Qh, mt * 16, HD, kt, lane);
        #pragma unroll
        for (int nt = 0; nt < 4; ++nt) s[mt][nt] = wmma_bf16(a, bfr[nt], s[mt][nt]);
      }
    }
    // + bias, softmax over j (row lives on 16 lanes x 4 N-tiles), write P bf16
    #pragma unroll
    for (int mt = 0; mt < 4; ++mt) {
      #pragma unroll
      for (int r = 0; r < 8; ++r) {
        int row = mt * 16 + r + rowoff;
        #pragma unroll
        for (int nt = 0; nt < 4; ++nt)
          s[mt][nt][r] += (float)Bh[row * KPTS + nt * 16 + n0];
        float mx = s[mt][0][r];
        #pragma unroll
        for (int nt = 1; nt < 4; ++nt) mx = fmaxf(mx, s[mt][nt][r]);
        #pragma unroll
        for (int msk = 8; msk >= 1; msk >>= 1) mx = fmaxf(mx, __shfl_xor(mx, msk, 16));
        float sum = 0.0f;
        #pragma unroll
        for (int nt = 0; nt < 4; ++nt) {
          float ev = __expf(s[mt][nt][r] - mx);
          s[mt][nt][r] = ev; sum += ev;
        }
        #pragma unroll
        for (int msk = 8; msk >= 1; msk >>= 1) sum += __shfl_xor(sum, msk, 16);
        float rs = 1.0f / sum;
        #pragma unroll
        for (int nt = 0; nt < 4; ++nt)
          Ph[row * KPTS + nt * 16 + n0] = (__bf16)(s[mt][nt][r] * rs);
      }
    }
    // O = P @ V  (V stored transposed so B-fragments are contiguous)
    v8f o[4][4] = {};
    #pragma unroll
    for (int kt = 0; kt < 2; ++kt) {
      bf16x16 bfr[4];
      #pragma unroll
      for (int nt = 0; nt < 4; ++nt) bfr[nt] = load_frag(Vh, nt * 16, KPTS, kt, lane);
      #pragma unroll
      for (int mt = 0; mt < 4; ++mt) {
        bf16x16 a = load_frag(Ph, mt * 16, KPTS, kt, lane);
        #pragma unroll
        for (int nt = 0; nt < 4; ++nt) o[mt][nt] = wmma_bf16(a, bfr[nt], o[mt][nt]);
      }
    }
    #pragma unroll
    for (int mt = 0; mt < 4; ++mt)
      #pragma unroll
      for (int nt = 0; nt < 4; ++nt)
        #pragma unroll
        for (int r = 0; r < 8; ++r)
          AGG[(mt * 16 + r + rowoff) * HID + hh * HD + nt * 16 + n0] = (__bf16)o[mt][nt][r];
  }
  __syncthreads();   // AGG complete; Q/K/Vt dead -> region D becomes Y

  // ---- phase 4: out projection + bias + residual -> Y f32 in LDS ----
  {
    const __bf16* AGG = HB;
    for (int ntp = 0; ntp < 2; ++ntp) {
      int nt0 = wave * 4 + ntp * 2;
      v8f acc[2][4] = {};
      for (int kt = 0; kt < KTH; ++kt) {
        if (kt + 1 < KTH)
          __builtin_prefetch(wo_frag + ((((kt + 1) * NTO + nt0) * 32 + lane) << 4), 0, 1);
        bf16x16 b0 = load_wfrag(wo_frag, NTO, kt, nt0,     lane);
        bf16x16 b1 = load_wfrag(wo_frag, NTO, kt, nt0 + 1, lane);
        #pragma unroll
        for (int mt = 0; mt < 4; ++mt) {
          bf16x16 a = load_frag(AGG, mt * 16, HID, kt, lane);
          acc[0][mt] = wmma_bf16(a, b0, acc[0][mt]);
          acc[1][mt] = wmma_bf16(a, b1, acc[1][mt]);
        }
      }
      #pragma unroll
      for (int u = 0; u < 2; ++u) {
        int c = (nt0 + u) * 16 + n0;
        float bo = b_out[c];
        #pragma unroll
        for (int mt = 0; mt < 4; ++mt)
          #pragma unroll
          for (int r = 0; r < 8; ++r) {
            int row = mt * 16 + r + rowoff;
            Y[row * HID + c] = acc[u][mt][r] + bo + hsrc[row * HID + c];
          }
      }
    }
  }
  __syncthreads();

  // ---- phase 5: LayerNorm (4 threads per row, shuffle-reduced) ----
  {
    int row  = tid >> 2;
    int part = tid & 3;
    const float* yr = Y + row * HID + part * 128;
    float s1 = 0.0f, s2 = 0.0f;
    for (int t = 0; t < 128; ++t) { float v = yr[t]; s1 += v; s2 += v * v; }
    s1 += __shfl_xor(s1, 1, 4); s1 += __shfl_xor(s1, 2, 4);
    s2 += __shfl_xor(s2, 1, 4); s2 += __shfl_xor(s2, 2, 4);
    float mean = s1 * (1.0f / 512.0f);
    float var  = s2 * (1.0f / 512.0f) - mean * mean;
    float rstd = rsqrtf(var + 1e-5f);
    float* orow = out + ((size_t)(mol * KPTS + row)) * HID + part * 128;
    const float* g = gamma + part * 128;
    const float* b = beta  + part * 128;
    for (int t = 0; t < 128; ++t)
      orow[t] = (yr[t] - mean) * rstd * g[t] + b[t];
  }
}

// ---------------------------------------------------------------------------
extern "C" void kernel_launch(void* const* d_in, const int* in_sizes, int n_in,
                              void* d_out, int out_size, void* d_ws, size_t ws_size,
                              hipStream_t stream) {
  const float* h       = (const float*)d_in[0];
  const float* pos     = (const float*)d_in[1];
  // d_in[2] = batch (unused: molecules are contiguous equal-size blocks)
  const float* centers = (const float*)d_in[3];
  const float* w_rbf   = (const float*)d_in[4];
  const float* b_rbf   = (const float*)d_in[5];
  const float* w_qkv   = (const float*)d_in[6];
  const float* b_qkv   = (const float*)d_in[7];
  const float* w_out   = (const float*)d_in[8];
  const float* b_out   = (const float*)d_in[9];
  const float* gamma   = (const float*)d_in[10];
  const float* beta    = (const float*)d_in[11];
  float* outp = (float*)d_out;

  __bf16* wq = (__bf16*)d_ws;                                  // 786432 bf16
  __bf16* wo = (__bf16*)((char*)d_ws + (size_t)786432 * 2);    // 262144 bf16

  const int total = 786432 + 262144;
  prep_weights<<<(total + 255) / 256, 256, 0, stream>>>(w_qkv, w_out, wq, wo);
  fused_attn<<<1024, 256, 0, stream>>>(h, pos, centers, w_rbf, b_rbf,
                                       wq, b_qkv, wo, b_out, gamma, beta, outp);
}
